// DPGNN_37958920962736
// MI455X (gfx1250) — matrix operand order repaired and checked by reading, not compiled, not run on hardware
//
#include <hip/hip_runtime.h>
#include <hip/hip_bf16.h>

// ---------------------------------------------------------------------------
// GraphSAGE (2x2 SAGEConv + MLP classifier) for MI455X (gfx1250, wave32).
// Phase 1 (edge scatter) is HBM-bandwidth bound -> bf16 gathers + f32 atomics.
// Phase 2 (dense) uses v_wmma_f32_16x16x32_bf16, one 16x16 tile per wave.
// ---------------------------------------------------------------------------

#define NN   100000   // nodes
#define CH   128      // in/hidden channels
#define OUTC 16       // classifier outputs

typedef __bf16 bf16_t;
typedef bf16_t bf16x16 __attribute__((ext_vector_type(16)));
typedef bf16_t bf16x8  __attribute__((ext_vector_type(8)));
typedef bf16_t bf16x4  __attribute__((ext_vector_type(4)));
typedef float  f32x8   __attribute__((ext_vector_type(8)));

// ------------------------------ small utils -------------------------------

__global__ void k_zero_f32(float* __restrict__ p, int n) {
  int i = blockIdx.x * blockDim.x + threadIdx.x;
  if (i < n) p[i] = 0.0f;
}

__global__ void k_f32_to_bf16(const float* __restrict__ in, bf16_t* __restrict__ out, int n) {
  int i = blockIdx.x * blockDim.x + threadIdx.x;
  if (i < n) out[i] = (bf16_t)in[i];
}

// degree count: one thread per edge
__global__ void k_count_deg(const int* __restrict__ dst, float* __restrict__ deg, int E) {
  int e = blockIdx.x * blockDim.x + threadIdx.x;
  if (e < E) atomicAdd(&deg[dst[e]], 1.0f);
}

// mean = sum / max(deg,1), emitted as bf16 for the WMMA stage
__global__ void k_finalize_mean(const float* __restrict__ sum, const float* __restrict__ deg,
                                bf16_t* __restrict__ out, int total) {
  int i = blockIdx.x * blockDim.x + threadIdx.x;
  if (i < total) {
    int n = i >> 7;                    // /CH
    float inv = 1.0f / fmaxf(deg[n], 1.0f);
    out[i] = (bf16_t)(sum[i] * inv);
  }
}

// ---------------------- edge scatter-add (wave per edge) ------------------
// Each wave handles one edge: 32 lanes x 4 channels = 128 channels.
// bf16 feature read (8B/lane) -> 4x global_atomic_add_f32 into the sum buffer.
__global__ void k_scatter_add(const bf16_t* __restrict__ feat,
                              const int* __restrict__ src,
                              const int* __restrict__ dst,
                              float* __restrict__ sum, int E) {
  int wave = blockIdx.x * (blockDim.x >> 5) + (threadIdx.x >> 5);
  if (wave >= E) return;
  int lane = threadIdx.x & 31;
  int s = src[wave];
  int d = dst[wave];
  const bf16_t* f = feat + (size_t)s * CH + lane * 4;
  float*        o = sum  + (size_t)d * CH + lane * 4;
  bf16x4 v = *(const bf16x4*)f;
  atomicAdd(o + 0, (float)v[0]);
  atomicAdd(o + 1, (float)v[1]);
  atomicAdd(o + 2, (float)v[2]);
  atomicAdd(o + 3, (float)v[3]);
}

// ----------------------------- WMMA fragments -----------------------------
// A (16x32 bf16, MxK): lane (m = lane&15), half = lane>=16.
//   VGPR 0..3 hold K = half*8 + {0..7}; VGPR 4..7 hold K = 16 + half*8 + {0..7}.
__device__ __forceinline__ bf16x16 load_a_frag(const bf16_t* __restrict__ arow, int ks, int half) {
  bf16x8 lo = *(const bf16x8*)(arow + ks + half * 8);
  bf16x8 hi = *(const bf16x8*)(arow + ks + 16 + half * 8);
  return __builtin_shufflevector(lo, hi, 0, 1, 2, 3, 4, 5, 6, 7, 8, 9, 10, 11, 12, 13, 14, 15);
}
// B (32x16 bf16, KxN): lane (n = lane&15) holds K = half*8 + {0..7}, then K = 16 + half*8 + {0..7}.
// We compute X @ W^T, so B[k][n] = W[n][k]: two 16B reads of W's row n.
__device__ __forceinline__ bf16x16 load_b_frag(const bf16_t* __restrict__ wrow, int ks, int half) {
  bf16x8 lo = *(const bf16x8*)(wrow + ks + half * 8);
  bf16x8 hi = *(const bf16x8*)(wrow + ks + 16 + half * 8);
  return __builtin_shufflevector(lo, hi, 0, 1, 2, 3, 4, 5, 6, 7, 8, 9, 10, 11, 12, 13, 14, 15);
}

// --------------------------- dual-A GEMM kernel ---------------------------
// D[m,n] = A0[m,:128] @ W0[n,:128]^T (+ A1 @ W1^T) + bias[n]; optional ReLU.
// One wave32 per 16x16 output tile; K=128 -> 4 (or 8) WMMA ops per tile.
__global__ __launch_bounds__(128)
void k_gemm_dual_bf16(const bf16_t* __restrict__ A0,
                      const bf16_t* __restrict__ A1,
                      const bf16_t* __restrict__ W0, int ldw0,
                      const bf16_t* __restrict__ W1, int ldw1,
                      const float* __restrict__ bias,
                      float* __restrict__ outF, int ncolsF,
                      bf16_t* __restrict__ outB,
                      int mtiles, int ntiles, int relu, int dualA) {
  int wave = blockIdx.x * (blockDim.x >> 5) + (threadIdx.x >> 5);
  if (wave >= mtiles * ntiles) return;
  int mt = wave / ntiles;
  int nt = wave - mt * ntiles;
  int lane = threadIdx.x & 31;
  int half = lane >> 4;
  int lr   = lane & 15;

  const bf16_t* a0row = A0 + (size_t)(mt * 16 + lr) * CH;
  const bf16_t* w0row = W0 + (size_t)(nt * 16 + lr) * ldw0;

  f32x8 c;
  float bv = bias ? bias[nt * 16 + lr] : 0.0f;
#pragma unroll
  for (int i = 0; i < 8; ++i) c[i] = bv;

#pragma unroll
  for (int ks = 0; ks < CH; ks += 32) {
    bf16x16 a = load_a_frag(a0row, ks, half);
    bf16x16 b = load_b_frag(w0row, ks, half);
    c = __builtin_amdgcn_wmma_f32_16x16x32_bf16(false, a, false, b, (short)0, c, false, false);
  }
  if (dualA) {
    const bf16_t* a1row = A1 + (size_t)(mt * 16 + lr) * CH;
    const bf16_t* w1row = W1 + (size_t)(nt * 16 + lr) * ldw1;
#pragma unroll
    for (int ks = 0; ks < CH; ks += 32) {
      bf16x16 a = load_a_frag(a1row, ks, half);
      bf16x16 b = load_b_frag(w1row, ks, half);
      c = __builtin_amdgcn_wmma_f32_16x16x32_bf16(false, a, false, b, (short)0, c, false, false);
    }
  }

  // C/D layout: VGPR v -> row (v + 8*half), col = lane&15.
  int n = nt * 16 + lr;
#pragma unroll
  for (int v = 0; v < 8; ++v) {
    float val = c[v];
    if (relu) val = fmaxf(val, 0.0f);
    size_t r = (size_t)(mt * 16 + v + half * 8);
    if (outF) outF[r * ncolsF + n] = val;
    if (outB) outB[r * CH + n] = (bf16_t)val;
  }
}

// ------------------------------- launcher ---------------------------------

static inline int cdiv(long long a, long long b) { return (int)((a + b - 1) / b); }

extern "C" void kernel_launch(void* const* d_in, const int* in_sizes, int n_in,
                              void* d_out, int out_size, void* d_ws, size_t ws_size,
                              hipStream_t stream) {
  const float* x      = (const float*)d_in[0];
  const int*   eidx   = (const int*)d_in[1];
  const int    E      = in_sizes[1] / 2;
  const int*   src    = eidx;
  const int*   dst    = eidx + E;

  const float* lc1_Wl = (const float*)d_in[2];
  const float* lc1_bl = (const float*)d_in[3];
  const float* lc1_Wr = (const float*)d_in[4];
  const float* lc2_Wl = (const float*)d_in[5];
  const float* lc2_bl = (const float*)d_in[6];
  const float* lc2_Wr = (const float*)d_in[7];
  const float* gc1_Wl = (const float*)d_in[8];
  const float* gc1_bl = (const float*)d_in[9];
  const float* gc1_Wr = (const float*)d_in[10];
  const float* gc2_Wl = (const float*)d_in[11];
  const float* gc2_bl = (const float*)d_in[12];
  const float* gc2_Wr = (const float*)d_in[13];
  const float* cls_W1 = (const float*)d_in[14];
  const float* cls_b1 = (const float*)d_in[15];
  const float* cls_W2 = (const float*)d_in[16];
  const float* cls_b2 = (const float*)d_in[17];

  // ---- workspace carve-out (~206 MB) ----
  char* base = (char*)d_ws;
  size_t off = 0;
  auto alloc = [&](size_t bytes) {
    void* r = base + off;
    off += (bytes + 255) & ~(size_t)255;
    return r;
  };
  float*  sumf   = (float*)alloc((size_t)NN * CH * 4); // f32 scatter accumulator
  float*  deg    = (float*)alloc((size_t)NN * 4);
  bf16_t* xb     = (bf16_t*)alloc((size_t)NN * CH * 2);
  bf16_t* meanxb = (bf16_t*)alloc((size_t)NN * CH * 2); // mean of x (shared l/g)
  bf16_t* meanb  = (bf16_t*)alloc((size_t)NN * CH * 2); // per-branch mean scratch
  bf16_t* hA     = (bf16_t*)alloc((size_t)NN * CH * 2); // layer-1 output (per branch)
  bf16_t* lb     = (bf16_t*)alloc((size_t)NN * CH * 2);
  bf16_t* gb     = (bf16_t*)alloc((size_t)NN * CH * 2);
  bf16_t* wl1    = (bf16_t*)alloc((size_t)CH * CH * 2);
  bf16_t* wr1    = (bf16_t*)alloc((size_t)CH * CH * 2);
  bf16_t* wl2    = (bf16_t*)alloc((size_t)CH * CH * 2);
  bf16_t* wr2    = (bf16_t*)alloc((size_t)CH * CH * 2);
  bf16_t* gwl1   = (bf16_t*)alloc((size_t)CH * CH * 2);
  bf16_t* gwr1   = (bf16_t*)alloc((size_t)CH * CH * 2);
  bf16_t* gwl2   = (bf16_t*)alloc((size_t)CH * CH * 2);
  bf16_t* gwr2   = (bf16_t*)alloc((size_t)CH * CH * 2);
  bf16_t* w1b    = (bf16_t*)alloc((size_t)CH * 2 * CH * 2); // [128,256]
  bf16_t* w2b    = (bf16_t*)alloc((size_t)OUTC * CH * 2);   // [16,128]
  bf16_t* clsh   = meanb; // classifier hidden reuses meanb (free by then)
  (void)ws_size; (void)n_in; (void)out_size;

  const int T = 256;
  const int nelF = NN * CH;                     // 12.8M
  const int wcnt = CH * CH;                     // 16384

  // ---- weight & input conversion to bf16 ----
  k_f32_to_bf16<<<cdiv(nelF, T), T, 0, stream>>>(x, xb, nelF);
  k_f32_to_bf16<<<cdiv(wcnt, T), T, 0, stream>>>(lc1_Wl, wl1, wcnt);
  k_f32_to_bf16<<<cdiv(wcnt, T), T, 0, stream>>>(lc1_Wr, wr1, wcnt);
  k_f32_to_bf16<<<cdiv(wcnt, T), T, 0, stream>>>(lc2_Wl, wl2, wcnt);
  k_f32_to_bf16<<<cdiv(wcnt, T), T, 0, stream>>>(lc2_Wr, wr2, wcnt);
  k_f32_to_bf16<<<cdiv(wcnt, T), T, 0, stream>>>(gc1_Wl, gwl1, wcnt);
  k_f32_to_bf16<<<cdiv(wcnt, T), T, 0, stream>>>(gc1_Wr, gwr1, wcnt);
  k_f32_to_bf16<<<cdiv(wcnt, T), T, 0, stream>>>(gc2_Wl, gwl2, wcnt);
  k_f32_to_bf16<<<cdiv(wcnt, T), T, 0, stream>>>(gc2_Wr, gwr2, wcnt);
  k_f32_to_bf16<<<cdiv(2 * wcnt, T), T, 0, stream>>>(cls_W1, w1b, 2 * wcnt);
  k_f32_to_bf16<<<cdiv(OUTC * CH, T), T, 0, stream>>>(cls_W2, w2b, OUTC * CH);

  // ---- degree counts (graph is fixed across layers) ----
  k_zero_f32<<<cdiv(NN, T), T, 0, stream>>>(deg, NN);
  k_count_deg<<<cdiv(E, T), T, 0, stream>>>(dst, deg, E);

  const int scatterBlocks = cdiv((long long)E * 32, T);
  const int mtiles = NN / 16;        // 6250
  const int ntHid = CH / 16;         // 8
  const int gHidBlocks = cdiv(mtiles * ntHid, 4); // 4 waves/block
  const int gOutBlocks = cdiv(mtiles * 1, 4);

  // ---- mean(x) once, shared by both branches ----
  k_zero_f32<<<cdiv(nelF, T), T, 0, stream>>>(sumf, nelF);
  k_scatter_add<<<scatterBlocks, T, 0, stream>>>(xb, src, dst, sumf, E);
  k_finalize_mean<<<cdiv(nelF, T), T, 0, stream>>>(sumf, deg, meanxb, nelF);

  // ---- l branch ----
  k_gemm_dual_bf16<<<gHidBlocks, 128, 0, stream>>>(meanxb, xb, wl1, CH, wr1, CH,
      lc1_bl, nullptr, 0, hA, mtiles, ntHid, /*relu=*/1, /*dual=*/1);
  k_zero_f32<<<cdiv(nelF, T), T, 0, stream>>>(sumf, nelF);
  k_scatter_add<<<scatterBlocks, T, 0, stream>>>(hA, src, dst, sumf, E);
  k_finalize_mean<<<cdiv(nelF, T), T, 0, stream>>>(sumf, deg, meanb, nelF);
  k_gemm_dual_bf16<<<gHidBlocks, 128, 0, stream>>>(meanb, hA, wl2, CH, wr2, CH,
      lc2_bl, nullptr, 0, lb, mtiles, ntHid, /*relu=*/0, /*dual=*/1);

  // ---- g branch (reuses hA / meanb) ----
  k_gemm_dual_bf16<<<gHidBlocks, 128, 0, stream>>>(meanxb, xb, gwl1, CH, gwr1, CH,
      gc1_bl, nullptr, 0, hA, mtiles, ntHid, /*relu=*/1, /*dual=*/1);
  k_zero_f32<<<cdiv(nelF, T), T, 0, stream>>>(sumf, nelF);
  k_scatter_add<<<scatterBlocks, T, 0, stream>>>(hA, src, dst, sumf, E);
  k_finalize_mean<<<cdiv(nelF, T), T, 0, stream>>>(sumf, deg, meanb, nelF);
  k_gemm_dual_bf16<<<gHidBlocks, 128, 0, stream>>>(meanb, hA, gwl2, CH, gwr2, CH,
      gc2_bl, nullptr, 0, gb, mtiles, ntHid, /*relu=*/0, /*dual=*/1);

  // ---- classifier: concat([l,g]) @ W1^T (split at col 128) -> relu -> W2^T ----
  k_gemm_dual_bf16<<<gHidBlocks, 128, 0, stream>>>(lb, gb, w1b, 2 * CH, w1b + CH, 2 * CH,
      cls_b1, nullptr, 0, clsh, mtiles, ntHid, /*relu=*/1, /*dual=*/1);
  k_gemm_dual_bf16<<<gOutBlocks, 128, 0, stream>>>(clsh, nullptr, w2b, CH, nullptr, 0,
      cls_b2, (float*)d_out, OUTC, nullptr, mtiles, OUTC / 16, /*relu=*/0, /*dual=*/0);
}
